// GraphEC_55748675502588
// MI455X (gfx1250) — matrix operand order repaired and checked by reading, not compiled, not run on hardware
//
#include <hip/hip_runtime.h>
#include <hip/hip_bf16.h>

typedef __bf16 bf16;
typedef bf16 v16bf __attribute__((ext_vector_type(16)));
typedef bf16 bf16x8 __attribute__((ext_vector_type(8)));
typedef float v8f __attribute__((ext_vector_type(8)));

#define ACT_NONE 0
#define ACT_SILU 1
#define ACT_TANH 2
#define ACT_SIGM 3
#define ACT_ELU  4

namespace {
constexpr int Nn   = 16000;
constexpr int Ne   = 192000;
constexpr int Bb   = 64;
constexpr int OUTD = 5069;
constexpr int OUTP = 5120;   // padded to 8 tiles of 16
}

// ---------------------------------------------------------------------------
// float <-> order-preserving uint (for atomic max on floats)
// ---------------------------------------------------------------------------
__device__ __forceinline__ unsigned f2ord(float f) {
  unsigned u = __float_as_uint(f);
  return (u & 0x80000000u) ? ~u : (u | 0x80000000u);
}
__device__ __forceinline__ float ord2f(unsigned o) {
  unsigned u = (o & 0x80000000u) ? (o & 0x7fffffffu) : ~o;
  return __uint_as_float(u);
}

__device__ __forceinline__ float act_eval(int ACT, float v) {
  if (ACT == ACT_SILU) return v / (1.f + __expf(-v));
  if (ACT == ACT_TANH) return tanhf(v);
  if (ACT == ACT_SIGM) return 1.f / (1.f + __expf(-v));
  if (ACT == ACT_ELU)  return v > 0.f ? v : __expf(v) - 1.f;
  return v;
}

// ---------------------------------------------------------------------------
// Weight repack: W[K,N] fp32 row-major -> wave32 WMMA B-fragment layout (bf16)
// element index: (((kt*NTtot + nt)*32 + lane)*16 + j)
//   K = kt*32 + (lane>>4)*8 + (j&7) + (j>>3)*16 ; N = nt*16 + (lane&15)
// mode 1: edge_mlp1 row permutation (concat reorder [h_r|h_c|radial|e] ->
//         [h_r|h_c|e|radial]) so every 32-K tile stays in one segment.
// ---------------------------------------------------------------------------
__global__ void pack_weights(const float* __restrict__ W, int K, int Nw,
                             bf16* __restrict__ Bp, int KP, int NP, int mode) {
  long t = (long)blockIdx.x * 256 + threadIdx.x;
  long total = (long)KP * NP;
  if (t >= total) return;
  int j    = t & 15;
  int lane = (t >> 4) & 31;
  long rest = t >> 9;
  int NT = NP >> 4;
  int nt = (int)(rest % NT);
  int kt = (int)(rest / NT);
  int k = kt * 32 + (lane >> 4) * 8 + (j & 7) + (j >> 3) * 16;
  int n = nt * 16 + (lane & 15);
  int ks = k;
  if (mode == 1) {
    if (k < 256)       ks = k;        // h[row] | h[col]
    else if (k < 384)  ks = k + 1;    // e  (orig rows 257..384)
    else if (k == 384) ks = 256;      // radial (orig row 256)
    else ks = -1;
  }
  float v = 0.f;
  if (ks >= 0 && ks < K && n < Nw) v = W[(long)ks * Nw + n];
  Bp[t] = (bf16)v;
}

// ---------------------------------------------------------------------------
// Generic WMMA GEMM: C[M,N] = act(A @ Bp + bias) (+resid)
// A: bf16 row-major, lda = padded K (multiple of 32).
// One wave computes a 16 x (16*NTILE) strip: A fragment loaded once per
// K-step; all NTILE B fragments preloaded (one long load clause + single
// wait), then NTILE back-to-back WMMAs.
// grid = (M/16, NPtiles/NTILE), block = 32.
// ---------------------------------------------------------------------------
template <int ACT, int NTILE>
__global__ __launch_bounds__(32)
void wmma_gemm(const bf16* __restrict__ A, long lda,
               const bf16* __restrict__ Bp, int KT, int NTtot,
               const float* __restrict__ bias,
               float* __restrict__ outF, long ldoF,
               const float* __restrict__ resid, long ldr,
               bf16* __restrict__ outB, long ldoB, int Nvalid) {
  const int lane = threadIdx.x & 31;
  const int mloc = lane & 15;
  const int half = lane >> 4;
  const long mBase = (long)blockIdx.x * 16;
  const int nt0 = blockIdx.y * NTILE;
  const bf16* arow = A + (mBase + mloc) * lda;
  v8f acc[NTILE];
#pragma unroll
  for (int q = 0; q < NTILE; ++q) acc[q] = {};
  for (int kt = 0; kt < KT; ++kt) {
    const int k0 = kt * 32;
    if (kt + 1 < KT) __builtin_prefetch(arow + k0 + 32, 0, 1);
    bf16x8 alo = *(const bf16x8*)(arow + k0 + half * 8);
    bf16x8 ahi = *(const bf16x8*)(arow + k0 + 16 + half * 8);
    const bf16* bprow = Bp + (((long)kt * NTtot + nt0) * 32 + lane) * 16;
    // preload all B fragments (single clause + single wait)
    bf16x8 blo[NTILE], bhi[NTILE];
#pragma unroll
    for (int q = 0; q < NTILE; ++q) {
      blo[q] = *(const bf16x8*)(bprow + (long)q * 512);
      bhi[q] = *(const bf16x8*)(bprow + (long)q * 512 + 8);
    }
    v16bf a;
#pragma unroll
    for (int i = 0; i < 8; ++i) { a[i] = alo[i]; a[8 + i] = ahi[i]; }
#pragma unroll
    for (int q = 0; q < NTILE; ++q) {
      v16bf b;
#pragma unroll
      for (int i = 0; i < 8; ++i) { b[i] = blo[q][i]; b[8 + i] = bhi[q][i]; }
      acc[q] = __builtin_amdgcn_wmma_f32_16x16x32_bf16(false, a, false, b,
                                                       (short)0, acc[q],
                                                       false, false);
    }
  }
#pragma unroll
  for (int q = 0; q < NTILE; ++q) {
    const int ncol = (nt0 + q) * 16 + mloc;
    const float bv = (bias && ncol < Nvalid) ? bias[ncol] : 0.f;
#pragma unroll
    for (int j = 0; j < 8; ++j) {
      long gr = mBase + j + 8 * half;
      float v = act_eval(ACT, acc[q][j] + bv);
      if (resid) v += resid[gr * ldr + ncol];
      if (ncol < Nvalid) {
        if (outF) outF[gr * ldoF + ncol] = v;
        if (outB) outB[gr * ldoB + ncol] = (bf16)v;
      }
    }
  }
}

// ---------------------------------------------------------------------------
// Edge-MLP1 GEMM with fused gather-concat A-loader.
// Virtual A[E,416] = [ h[row] (0..127) | h[col] (128..255) | e (256..383) |
//                      radial (384) | zeros ].  SiLU epilogue.
// One wave computes a full 16x128 strip (gathered A loaded once per K-step,
// 8 B fragments preloaded, 8 back-to-back WMMAs).  grid = (E/16, 1).
// ---------------------------------------------------------------------------
__global__ __launch_bounds__(32)
void wmma_gemm_edge(const int* __restrict__ row, const int* __restrict__ col,
                    const bf16* __restrict__ hB, const bf16* __restrict__ eOld,
                    const bf16* __restrict__ radB,
                    const bf16* __restrict__ Bp, const float* __restrict__ bias,
                    bf16* __restrict__ outB) {
  const int lane = threadIdx.x & 31;
  const int mloc = lane & 15;
  const int half = lane >> 4;
  const long m = (long)blockIdx.x * 16 + mloc;
  const int r = row[m];
  const int c = col[m];
  v8f acc[8];
#pragma unroll
  for (int q = 0; q < 8; ++q) acc[q] = {};
  for (int kt = 0; kt < 13; ++kt) {
    const int k0 = kt * 32;
    const bf16* bprow = Bp + ((long)kt * 8 * 32 + lane) * 16;
    bf16x8 blo[8], bhi[8];
#pragma unroll
    for (int q = 0; q < 8; ++q) {
      blo[q] = *(const bf16x8*)(bprow + (long)q * 512);
      bhi[q] = *(const bf16x8*)(bprow + (long)q * 512 + 8);
    }
    v16bf a;
    if (k0 < 384) {
      const bf16* base = (k0 < 128) ? (hB + (long)r * 128)
                       : (k0 < 256) ? (hB + (long)c * 128)
                                    : (eOld + m * 128);
      const int off = (k0 & 127) + half * 8;
      bf16x8 alo = *(const bf16x8*)(base + off);
      bf16x8 ahi = *(const bf16x8*)(base + off + 16);
#pragma unroll
      for (int i = 0; i < 8; ++i) { a[i] = alo[i]; a[8 + i] = ahi[i]; }
    } else {
#pragma unroll
      for (int i = 0; i < 16; ++i) a[i] = (bf16)0.f;
      if (half == 0) a[0] = radB[m];   // K == 384
    }
#pragma unroll
    for (int q = 0; q < 8; ++q) {
      v16bf b;
#pragma unroll
      for (int i = 0; i < 8; ++i) { b[i] = blo[q][i]; b[8 + i] = bhi[q][i]; }
      acc[q] = __builtin_amdgcn_wmma_f32_16x16x32_bf16(false, a, false, b,
                                                       (short)0, acc[q],
                                                       false, false);
    }
  }
#pragma unroll
  for (int q = 0; q < 8; ++q) {
    const int ncol = q * 16 + mloc;
    const float bv = bias[ncol];
#pragma unroll
    for (int j = 0; j < 8; ++j) {
      long gr = (long)blockIdx.x * 16 + j + 8 * half;
      float v = acc[q][j] + bv;
      v = v / (1.f + __expf(-v));               // SiLU
      outB[gr * 128 + ncol] = (bf16)v;
    }
  }
}

// ---------------------------------------------------------------------------
// Elementwise / prep kernels
// ---------------------------------------------------------------------------
__global__ void cvt_pad(const float* __restrict__ src, bf16* __restrict__ dst,
                        long M, int K, int KP) {
  long t = (long)blockIdx.x * 256 + threadIdx.x;
  if (t >= M * (long)KP) return;
  long m = t / KP; int k = (int)(t % KP);
  dst[t] = (k < K) ? (bf16)src[m * K + k] : (bf16)0.f;
}

__global__ void node_prep(const float* __restrict__ X, float* __restrict__ xcur,
                          bf16* __restrict__ sfeatB) {
  int n = blockIdx.x * 256 + threadIdx.x;
  if (n >= Nn) return;
  const int np = (n + 1) % Nn, nm = (n - 1 + Nn) % Nn;
  float c0 = X[(long)n * 12 + 3], c1 = X[(long)n * 12 + 4], c2 = X[(long)n * 12 + 5];
  float f0 = X[(long)np * 12 + 3] - c0, f1 = X[(long)np * 12 + 4] - c1, f2 = X[(long)np * 12 + 5] - c2;
  float b0 = X[(long)nm * 12 + 3] - c0, b1 = X[(long)nm * 12 + 4] - c1, b2 = X[(long)nm * 12 + 5] - c2;
  float fl = sqrtf(f0 * f0 + f1 * f1 + f2 * f2) + 1e-6f;
  float bl = sqrtf(b0 * b0 + b1 * b1 + b2 * b2) + 1e-6f;
  f0 /= fl; f1 /= fl; f2 /= fl;
  b0 /= bl; b1 /= bl; b2 /= bl;
  float x0 = b1 * f2 - b2 * f1, x1 = b2 * f0 - b0 * f2, x2 = b0 * f1 - b1 * f0;
  bf16* g = sfeatB + (long)n * 192 + 175;
  g[0] = (bf16)f0; g[1] = (bf16)f1; g[2] = (bf16)f2;
  g[3] = (bf16)b0; g[4] = (bf16)b1; g[5] = (bf16)b2;
  g[6] = (bf16)x0; g[7] = (bf16)x1; g[8] = (bf16)x2;
  xcur[n * 3 + 0] = c0; xcur[n * 3 + 1] = c1; xcur[n * 3 + 2] = c2;
}

__global__ void edge_prep(const float* __restrict__ X, const int* __restrict__ ei,
                          bf16* __restrict__ hEB, float* __restrict__ deg) {
  int e = blockIdx.x * 256 + threadIdx.x;
  if (e >= Ne) return;
  int r = ei[e], c = ei[Ne + e];
  float d0 = X[(long)r * 12 + 3] - X[(long)c * 12 + 3];
  float d1 = X[(long)r * 12 + 4] - X[(long)c * 12 + 4];
  float d2 = X[(long)r * 12 + 5] - X[(long)c * 12 + 5];
  float d = sqrtf(d0 * d0 + d1 * d1 + d2 * d2);
  bf16* o = hEB + (long)e * 32;
#pragma unroll
  for (int j = 0; j < 16; ++j) {
    float mu = (20.f / 15.f) * j;
    float z = (d - mu) / 1.25f;
    o[j] = (bf16)__expf(-z * z);
  }
#pragma unroll
  for (int j = 16; j < 32; ++j) o[j] = (bf16)0.f;
  atomicAdd(&deg[r], 1.f);
}

__global__ void edge_radial_k(const float* __restrict__ xcur, const int* __restrict__ ei,
                              bf16* __restrict__ radB, float* __restrict__ diffN) {
  int e = blockIdx.x * 256 + threadIdx.x;
  if (e >= Ne) return;
  int r = ei[e], c = ei[Ne + e];
  float d0 = xcur[r * 3 + 0] - xcur[c * 3 + 0];
  float d1 = xcur[r * 3 + 1] - xcur[c * 3 + 1];
  float d2 = xcur[r * 3 + 2] - xcur[c * 3 + 2];
  float rad = d0 * d0 + d1 * d1 + d2 * d2;
  float inv = 1.f / (sqrtf(rad) + 1.f);
  radB[e] = (bf16)rad;
  diffN[(long)e * 3 + 0] = d0 * inv;
  diffN[(long)e * 3 + 1] = d1 * inv;
  diffN[(long)e * 3 + 2] = d2 * inv;
}

// m = m2 * sigmoid(att); write gated m into eB (becomes next-layer e) and
// scatter-accumulate agg[row] += m.
__global__ void edge_gate(const bf16* __restrict__ m2B, const float* __restrict__ attP,
                          const int* __restrict__ ei, bf16* __restrict__ eB,
                          float* __restrict__ aggF) {
  long t = (long)blockIdx.x * 256 + threadIdx.x;
  if (t >= (long)Ne * 128) return;
  long e = t >> 7; int k = (int)(t & 127);
  float v = (float)m2B[t] * attP[e];
  eB[t] = (bf16)v;
  atomicAdd(&aggF[(long)ei[e] * 128 + k], v);
}

__global__ void coord_scatter(const int* __restrict__ ei, const float* __restrict__ tP,
                              const float* __restrict__ diffN, float* __restrict__ transAcc) {
  int e = blockIdx.x * 256 + threadIdx.x;
  if (e >= Ne) return;
  float t = tP[e];
  int r = ei[e];
  atomicAdd(&transAcc[r * 3 + 0], diffN[(long)e * 3 + 0] * t);
  atomicAdd(&transAcc[r * 3 + 1], diffN[(long)e * 3 + 1] * t);
  atomicAdd(&transAcc[r * 3 + 2], diffN[(long)e * 3 + 2] * t);
}

__global__ void coord_apply(float* __restrict__ xcur, const float* __restrict__ transAcc,
                            const float* __restrict__ deg) {
  int n = blockIdx.x * 256 + threadIdx.x;
  if (n >= Nn) return;
  float inv = 1.f / fmaxf(deg[n], 1.f);
  xcur[n * 3 + 0] += transAcc[n * 3 + 0] * inv;
  xcur[n * 3 + 1] += transAcc[n * 3 + 1] * inv;
  xcur[n * 3 + 2] += transAcc[n * 3 + 2] * inv;
}

__global__ void node_concat(const bf16* __restrict__ hB, const float* __restrict__ aggF,
                            bf16* __restrict__ nodeInB) {
  long t = (long)blockIdx.x * 256 + threadIdx.x;
  if (t >= (long)Nn * 256) return;
  long n = t >> 8; int k = (int)(t & 255);
  nodeInB[t] = (k < 128) ? hB[n * 128 + k] : (bf16)aggF[n * 128 + (k - 128)];
}

__global__ void mx_init(unsigned* __restrict__ mxOrd) {
  int t = threadIdx.x;
  if (t < Bb * 4) mxOrd[t] = f2ord(-3.0e38f);
}

__global__ void attn_small(const float* __restrict__ attHid, const float* __restrict__ W2,
                           const float* __restrict__ b2, const int* __restrict__ bid,
                           float* __restrict__ logitsF, unsigned* __restrict__ mxOrd) {
  int n = blockIdx.x * 256 + threadIdx.x;
  if (n >= Nn) return;
  float h[16];
#pragma unroll
  for (int j = 0; j < 16; ++j) h[j] = attHid[(long)n * 16 + j];
  int b = bid[n];
#pragma unroll
  for (int o = 0; o < 4; ++o) {
    float l = b2[o];
#pragma unroll
    for (int j = 0; j < 16; ++j) l += h[j] * W2[j * 4 + o];
    logitsF[(long)n * 4 + o] = l;
    atomicMax(&mxOrd[b * 4 + o], f2ord(l));
  }
}

__global__ void attn_w(const float* __restrict__ logitsF, const unsigned* __restrict__ mxOrd,
                       const int* __restrict__ bid, float* __restrict__ wFbuf,
                       float* __restrict__ denomF) {
  int n = blockIdx.x * 256 + threadIdx.x;
  if (n >= Nn) return;
  int b = bid[n];
#pragma unroll
  for (int o = 0; o < 4; ++o) {
    float w = __expf(logitsF[(long)n * 4 + o] - ord2f(mxOrd[b * 4 + o]));
    wFbuf[(long)n * 4 + o] = w;
    atomicAdd(&denomF[b * 4 + o], w);
  }
}

__global__ void attn_wt(const float* __restrict__ wFbuf, const float* __restrict__ denomF,
                        const int* __restrict__ bid, float* __restrict__ wtF) {
  int n = blockIdx.x * 256 + threadIdx.x;
  if (n >= Nn) return;
  int b = bid[n];
  float s = 0.f;
#pragma unroll
  for (int o = 0; o < 4; ++o) s += wFbuf[(long)n * 4 + o] / denomF[b * 4 + o];
  wtF[n] = s;
}

__global__ void pool_acc(const bf16* __restrict__ embB, const float* __restrict__ wtF,
                         const int* __restrict__ bid, float* __restrict__ pooledF) {
  long t = (long)blockIdx.x * 256 + threadIdx.x;
  if (t >= (long)Nn * 256) return;
  long n = t >> 8; int c = (int)(t & 255);
  atomicAdd(&pooledF[(long)bid[n] * 256 + c], wtF[n] * (float)embB[t]);
}

// ---------------------------------------------------------------------------
// Host launcher
// ---------------------------------------------------------------------------
extern "C" void kernel_launch(void* const* d_in, const int* in_sizes, int n_in,
                              void* d_out, int out_size, void* d_ws, size_t ws_size,
                              hipStream_t stream) {
  (void)in_sizes; (void)n_in; (void)out_size; (void)ws_size;
  const float* X       = (const float*)d_in[0];
  const float* structf = (const float*)d_in[1];
  const float* seqf    = (const float*)d_in[2];
  const int*   eidx    = (const int*)d_in[3];
  const int*   bid     = (const int*)d_in[4];
  const float* W_emb_in = (const float*)d_in[5];
  const float* b_emb_in = (const float*)d_in[6];
  const float* W_ee_in  = (const float*)d_in[7];
  const float* b_ee_in  = (const float*)d_in[8];
  const float *Wl[2][7], *bl[2][7];
  for (int l = 0; l < 2; ++l) {
    int base = 9 + 13 * l;
    Wl[l][0] = (const float*)d_in[base + 0];  bl[l][0] = (const float*)d_in[base + 1];   // edge_mlp1
    Wl[l][1] = (const float*)d_in[base + 2];  bl[l][1] = (const float*)d_in[base + 3];   // edge_mlp2
    Wl[l][2] = (const float*)d_in[base + 4];  bl[l][2] = (const float*)d_in[base + 5];   // att_mlp
    Wl[l][3] = (const float*)d_in[base + 6];  bl[l][3] = (const float*)d_in[base + 7];   // coord_mlp1
    Wl[l][4] = (const float*)d_in[base + 8];  bl[l][4] = nullptr;                        // coord_mlp2 (no bias)
    Wl[l][5] = (const float*)d_in[base + 9];  bl[l][5] = (const float*)d_in[base + 10];  // node_mlp1
    Wl[l][6] = (const float*)d_in[base + 11]; bl[l][6] = (const float*)d_in[base + 12];  // node_mlp2
  }
  const float* W_emb_out = (const float*)d_in[35];
  const float* b_emb_out = (const float*)d_in[36];
  const float* W_seq  = (const float*)d_in[39];
  const float* b_seq  = (const float*)d_in[40];
  const float* W_fc1  = (const float*)d_in[41];
  const float* b_fc1  = (const float*)d_in[42];
  const float* W_fc2  = (const float*)d_in[43];
  const float* b_fc2  = (const float*)d_in[44];
  const float* W_proj = (const float*)d_in[45];
  const float* b_proj = (const float*)d_in[46];
  const float* W_out  = (const float*)d_in[47];
  const float* b_out  = (const float*)d_in[48];

  // ---- workspace carve-out (256B aligned) ----
  char* p = (char*)d_ws;
  auto alloc = [&](size_t bytes) -> char* {
    char* r = p; p += (bytes + 255) & ~(size_t)255; return r;
  };
  float* xcur    = (float*)alloc((size_t)Nn * 3 * 4);
  float* deg     = (float*)alloc((size_t)Nn * 4);
  bf16*  sfeatB  = (bf16*) alloc((size_t)Nn * 192 * 2);
  bf16*  seqB    = (bf16*) alloc((size_t)Nn * 1056 * 2);
  float* hF      = (float*)alloc((size_t)Nn * 128 * 4);
  bf16*  hB      = (bf16*) alloc((size_t)Nn * 128 * 2);
  bf16*  eB      = (bf16*) alloc((size_t)Ne * 128 * 2);
  bf16*  hEB     = (bf16*) alloc((size_t)Ne * 32 * 2);
  bf16*  radB    = (bf16*) alloc((size_t)Ne * 2);
  float* diffN   = (float*)alloc((size_t)Ne * 3 * 4);
  bf16*  m1B     = (bf16*) alloc((size_t)Ne * 128 * 2);
  bf16*  m2B     = (bf16*) alloc((size_t)Ne * 128 * 2);
  float* attP    = (float*)alloc((size_t)Ne * 4);
  bf16*  c1B     = (bf16*) alloc((size_t)Ne * 128 * 2);
  float* tP      = (float*)alloc((size_t)Ne * 4);
  float* aggF    = (float*)alloc((size_t)Nn * 128 * 4);
  float* transAcc= (float*)alloc((size_t)Nn * 3 * 4);
  bf16*  nodeInB = (bf16*) alloc((size_t)Nn * 256 * 2);
  bf16*  n1B     = (bf16*) alloc((size_t)Nn * 128 * 2);
  bf16*  embB    = (bf16*) alloc((size_t)Nn * 256 * 2);
  float* attHid  = (float*)alloc((size_t)Nn * 16 * 4);
  float* logitsF = (float*)alloc((size_t)Nn * 4 * 4);
  float* wFbuf   = (float*)alloc((size_t)Nn * 4 * 4);
  float* wtF     = (float*)alloc((size_t)Nn * 4);
  unsigned* mxOrd= (unsigned*)alloc((size_t)Bb * 4 * 4);
  float* denomF  = (float*)alloc((size_t)Bb * 4 * 4);
  float* pooledF = (float*)alloc((size_t)Bb * 256 * 4);
  bf16*  pooledB = (bf16*) alloc((size_t)Bb * 256 * 2);
  bf16*  projB   = (bf16*) alloc((size_t)Bb * 128 * 2);
  // packed weights
  bf16* Pemb_in  = (bf16*)alloc((size_t)192 * 128 * 2);
  bf16* Pee_in   = (bf16*)alloc((size_t)32 * 128 * 2);
  bf16 *Pe1[2], *Pe2[2], *Patt[2], *Pc1[2], *Pc2[2], *Pn1[2], *Pn2[2];
  for (int l = 0; l < 2; ++l) {
    Pe1[l]  = (bf16*)alloc((size_t)416 * 128 * 2);
    Pe2[l]  = (bf16*)alloc((size_t)128 * 128 * 2);
    Patt[l] = (bf16*)alloc((size_t)128 * 16 * 2);
    Pc1[l]  = (bf16*)alloc((size_t)128 * 128 * 2);
    Pc2[l]  = (bf16*)alloc((size_t)128 * 16 * 2);
    Pn1[l]  = (bf16*)alloc((size_t)256 * 128 * 2);
    Pn2[l]  = (bf16*)alloc((size_t)128 * 128 * 2);
  }
  bf16* Pemb_out = (bf16*)alloc((size_t)128 * 128 * 2);
  bf16* Pseq     = (bf16*)alloc((size_t)1056 * 128 * 2);
  bf16* Pfc1     = (bf16*)alloc((size_t)256 * 16 * 2);
  bf16* Pproj    = (bf16*)alloc((size_t)256 * 128 * 2);
  bf16* Pout     = (bf16*)alloc((size_t)128 * OUTP * 2);

  auto packL = [&](const float* W, int K, int Nw, bf16* Bp, int KP, int NP, int mode) {
    long total = (long)KP * NP;
    pack_weights<<<(int)((total + 255) / 256), 256, 0, stream>>>(W, K, Nw, Bp, KP, NP, mode);
  };
  packL(W_emb_in, 184, 128, Pemb_in, 192, 128, 0);
  packL(W_ee_in,  16,  128, Pee_in,  32,  128, 0);
  for (int l = 0; l < 2; ++l) {
    packL(Wl[l][0], 385, 128, Pe1[l], 416, 128, 1);   // permuted concat
    packL(Wl[l][1], 128, 128, Pe2[l], 128, 128, 0);
    packL(Wl[l][2], 128, 1,   Patt[l],128, 16,  0);
    packL(Wl[l][3], 128, 128, Pc1[l], 128, 128, 0);
    packL(Wl[l][4], 128, 1,   Pc2[l], 128, 16,  0);
    packL(Wl[l][5], 256, 128, Pn1[l], 256, 128, 0);
    packL(Wl[l][6], 128, 128, Pn2[l], 128, 128, 0);
  }
  packL(W_emb_out, 128, 128, Pemb_out, 128, 128, 0);
  packL(W_seq, 1036, 128, Pseq, 1056, 128, 0);
  packL(W_fc1, 256, 16, Pfc1, 256, 16, 0);
  packL(W_proj, 256, 128, Pproj, 256, 128, 0);
  packL(W_out, 128, OUTD, Pout, 128, OUTP, 0);

  // NTILE=8 strip GEMM (grid.y = NTtot/8) and NTILE=1 narrow GEMM
#define GEMM8(ACT, A, lda, Bp, KT, MT, NTtot, bias, oF, ldoF, res, ldr, oB, ldoB, Nv) \
  wmma_gemm<ACT, 8><<<dim3((MT), (NTtot) / 8), 32, 0, stream>>>(                       \
      (A), (lda), (Bp), (KT), (NTtot), (bias), (oF), (ldoF), (res), (ldr), (oB), (ldoB), (Nv))
#define GEMM1(ACT, A, lda, Bp, KT, MT, bias, oF, ldoF, Nv)                             \
  wmma_gemm<ACT, 1><<<dim3((MT), 1), 32, 0, stream>>>(                                 \
      (A), (lda), (Bp), (KT), 1, (bias), (oF), (ldoF), nullptr, 0, (bf16*)nullptr, 0, (Nv))

  const int MTn = Nn / 16;   // 1000
  const int MTe = Ne / 16;   // 12000

  // ---- prep ----
  hipMemsetAsync(deg, 0, (size_t)Nn * 4, stream);
  cvt_pad<<<(int)(((long)Nn * 192 + 255) / 256), 256, 0, stream>>>(structf, sfeatB, Nn, 175, 192);
  cvt_pad<<<(int)(((long)Nn * 1056 + 255) / 256), 256, 0, stream>>>(seqf, seqB, Nn, 1036, 1056);
  node_prep<<<(Nn + 255) / 256, 256, 0, stream>>>(X, xcur, sfeatB);
  edge_prep<<<(Ne + 255) / 256, 256, 0, stream>>>(X, eidx, hEB, deg);

  // ---- input embeddings ----
  GEMM8(ACT_NONE, sfeatB, 192, Pemb_in, 6, MTn, 8, b_emb_in, hF, 128, nullptr, 0, hB, 128, 128);
  GEMM8(ACT_NONE, hEB, 32, Pee_in, 1, MTe, 8, b_ee_in, (float*)nullptr, 0, nullptr, 0, eB, 128, 128);

  // ---- EGNN layers ----
  for (int l = 0; l < 2; ++l) {
    edge_radial_k<<<(Ne + 255) / 256, 256, 0, stream>>>(xcur, eidx, radB, diffN);
    wmma_gemm_edge<<<dim3(MTe, 1), 32, 0, stream>>>(eidx, eidx + Ne, hB, eB, radB,
                                                    Pe1[l], bl[l][0], m1B);
    GEMM8(ACT_SILU, m1B, 128, Pe2[l], 4, MTe, 8, bl[l][1], (float*)nullptr, 0, nullptr, 0, m2B, 128, 128);
    GEMM1(ACT_SIGM, m2B, 128, Patt[l], 4, MTe, bl[l][2], attP, 1, 1);
    hipMemsetAsync(aggF, 0, (size_t)Nn * 128 * 4, stream);
    edge_gate<<<(int)(((long)Ne * 128 + 255) / 256), 256, 0, stream>>>(m2B, attP, eidx, eB, aggF);
    GEMM8(ACT_SILU, eB, 128, Pc1[l], 4, MTe, 8, bl[l][3], (float*)nullptr, 0, nullptr, 0, c1B, 128, 128);
    GEMM1(ACT_TANH, c1B, 128, Pc2[l], 4, MTe, (const float*)nullptr, tP, 1, 1);
    hipMemsetAsync(transAcc, 0, (size_t)Nn * 3 * 4, stream);
    coord_scatter<<<(Ne + 255) / 256, 256, 0, stream>>>(eidx, tP, diffN, transAcc);
    coord_apply<<<(Nn + 255) / 256, 256, 0, stream>>>(xcur, transAcc, deg);
    node_concat<<<(int)(((long)Nn * 256 + 255) / 256), 256, 0, stream>>>(hB, aggF, nodeInB);
    GEMM8(ACT_SILU, nodeInB, 256, Pn1[l], 8, MTn, 8, bl[l][5], (float*)nullptr, 0, nullptr, 0, n1B, 128, 128);
    GEMM8(ACT_NONE, n1B, 128, Pn2[l], 4, MTn, 8, bl[l][6], hF, 128, hF, 128, hB, 128, 128);
  }

  // ---- head: emb = [emb_out(h) | seq_d1(seq)] ----
  GEMM8(ACT_NONE, hB, 128, Pemb_out, 4, MTn, 8, b_emb_out, (float*)nullptr, 0, nullptr, 0, embB, 256, 128);
  GEMM8(ACT_NONE, seqB, 1056, Pseq, 33, MTn, 8, b_seq, (float*)nullptr, 0, nullptr, 0, embB + 128, 256, 128);
  GEMM1(ACT_TANH, embB, 256, Pfc1, 8, MTn, b_fc1, attHid, 16, 16);
  mx_init<<<1, 256, 0, stream>>>(mxOrd);
  attn_small<<<(Nn + 255) / 256, 256, 0, stream>>>(attHid, W_fc2, b_fc2, bid, logitsF, mxOrd);
  hipMemsetAsync(denomF, 0, (size_t)Bb * 4 * 4, stream);
  attn_w<<<(Nn + 255) / 256, 256, 0, stream>>>(logitsF, mxOrd, bid, wFbuf, denomF);
  attn_wt<<<(Nn + 255) / 256, 256, 0, stream>>>(wFbuf, denomF, bid, wtF);
  hipMemsetAsync(pooledF, 0, (size_t)Bb * 256 * 4, stream);
  pool_acc<<<(int)(((long)Nn * 256 + 255) / 256), 256, 0, stream>>>(embB, wtF, bid, pooledF);
  cvt_pad<<<(Bb * 256 + 255) / 256, 256, 0, stream>>>(pooledF, pooledB, Bb, 256, 256);
  GEMM8(ACT_ELU, pooledB, 256, Pproj, 8, Bb / 16, 8, b_proj, (float*)nullptr, 0, nullptr, 0, projB, 128, 128);
  GEMM8(ACT_NONE, projB, 128, Pout, 4, Bb / 16, OUTP / 16, b_out, (float*)d_out, OUTD, nullptr, 0, (bf16*)nullptr, 0, OUTD);
#undef GEMM8
#undef GEMM1
}
